// SimpleGCNModel_38362647888477
// MI455X (gfx1250) — compile-verified, matrix operand order checked by reading, not run on hardware
//
#include <hip/hip_runtime.h>

// ---------------------------------------------------------------------------
// Types for CDNA5 WMMA (wave32)
// ---------------------------------------------------------------------------
typedef __attribute__((ext_vector_type(16))) __bf16 bf16x16;
typedef __attribute__((ext_vector_type(8)))  __bf16 bf16x8;
typedef __attribute__((ext_vector_type(2)))  __bf16 bf16x2;
typedef __attribute__((ext_vector_type(8)))  float  f32x8;
typedef __attribute__((ext_vector_type(2)))  float  f32x2;

#define BM 128
#define BK 32
#define LDS_STRIDE 40   // bf16 elems per row (80B rows: 16B aligned, spreads banks)

// Async global->LDS: one 64B row segment as 4x b128 off a single base address.
// ISA: the instruction offset applies to both the LDS and global address -> the literal
// offset advances BOTH addresses, so one address setup serves all 4 chunks.
__device__ __forceinline__ void async_copy_row64(const void* gsrc, void* ldst)
{
    unsigned lds_off = (unsigned)(size_t)ldst;   // generic->LDS = truncate
    asm volatile("global_load_async_to_lds_b128 %0, %1, off"           :: "v"(lds_off), "v"(gsrc) : "memory");
    asm volatile("global_load_async_to_lds_b128 %0, %1, off offset:16" :: "v"(lds_off), "v"(gsrc) : "memory");
    asm volatile("global_load_async_to_lds_b128 %0, %1, off offset:32" :: "v"(lds_off), "v"(gsrc) : "memory");
    asm volatile("global_load_async_to_lds_b128 %0, %1, off offset:48" :: "v"(lds_off), "v"(gsrc) : "memory");
}
__device__ __forceinline__ void wait_async_lds()
{
    asm volatile("s_wait_asynccnt 0x0" ::: "memory");
}

// ---------------------------------------------------------------------------
// Edge scatter-add: agg[dst[e], :] += x_src[src[e], :]
// ---------------------------------------------------------------------------
__global__ __launch_bounds__(256)
void gcn_scatter_add(const float* __restrict__ xsrc,
                     const int* __restrict__ src,
                     const int* __restrict__ dst,
                     float* __restrict__ agg, int D)
{
    const int e = blockIdx.x;
    const int s = src[e];
    const int d = dst[e];
    const float* __restrict__ row = xsrc + (size_t)s * D;
    float* __restrict__ out = agg + (size_t)d * D;
    for (int i = threadIdx.x; i < D; i += 256)
        atomicAdd(&out[i], row[i]);
}

// ---------------------------------------------------------------------------
// Weight prep (once per launch): W [K][N] f32 -> hi/lo bf16 planes [N][K]
// ---------------------------------------------------------------------------
__global__ __launch_bounds__(256)
void prep_weight(const float* __restrict__ W, __bf16* __restrict__ hi,
                 __bf16* __restrict__ lo, int K, int N)
{
    int idx = blockIdx.x * 256 + threadIdx.x;
    if (idx >= K * N) return;
    int k = idx / N, n = idx % N;       // coalesced read of W
    float x = W[idx];
    __bf16 h = (__bf16)x;
    hi[(size_t)n * K + k] = h;
    lo[(size_t)n * K + k] = (__bf16)(x - (float)h);
}

// ---------------------------------------------------------------------------
// Split-bf16 GEMM, double-buffered LDS pipeline, optional fused second pair:
//   OUT[m,n] = act( A0 @ B0^T (+ A1 @ B1^T) + bias[n] )
// A: f32 [M][K], converted to hi/lo bf16 planes in LDS on the fly
//    (v_cvt_pk_bf16_f32 + v_fma_mix residual).
// B: pre-split bf16 planes [N][K], staged with global_load_async_to_lds_b128.
// Block tile 128 x (NT*32); 8 waves = 4(M) x 2(N); wave tile 32 x NT*16.
// 3 WMMAs per 16x16x32 step: hi*hi + hi*lo + lo*hi  (~fp32 accuracy).
// Pipeline: one barrier per K-step; the K-loop is unrolled by two with
// literal ping-pong buffer indices and the last two steps peeled, so the
// steady-state loop stages unconditionally with immediate LDS offsets.
// ---------------------------------------------------------------------------
template <int NT>
__global__ __launch_bounds__(256)
void gemm_bf16split(const float* __restrict__ A0,
                    const __bf16* __restrict__ B0hi, const __bf16* __restrict__ B0lo, int K0,
                    const float* __restrict__ A1,
                    const __bf16* __restrict__ B1hi, const __bf16* __restrict__ B1lo, int K1,
                    const float* __restrict__ bias, float* __restrict__ OUT,
                    int ldo, int do_relu)
{
    constexpr int BN_ = NT * 32;

    __shared__ __bf16 sAhi[2][BM  * LDS_STRIDE];
    __shared__ __bf16 sAlo[2][BM  * LDS_STRIDE];
    __shared__ __bf16 sBhi[2][BN_ * LDS_STRIDE];   // row = n, col = k
    __shared__ __bf16 sBlo[2][BN_ * LDS_STRIDE];

    const int tid  = threadIdx.x;
    const int lane = tid & 31;
    const int wave = tid >> 5;
    const int waveM = wave & 3;       // 4 waves along M, 32 rows each
    const int waveN = wave >> 2;      // 2 waves along N, NT*16 cols each
    const int kh   = lane >> 4;
    const int l16  = lane & 15;

    const int blockM = blockIdx.x * BM;
    const int blockN = blockIdx.y * BN_;

    // fixed per-thread staging coordinates
    const int aRow = tid >> 4;             // rows aRow + 16*j, j<8
    const int aCol = (tid & 15) * 2;       // fixed k-pair column
    const bool bActive = (NT == 4) || (tid < BN_ * 2);
    const int bRow = tid >> 1;             // one 64B segment per thread
    const int bPl  = tid & 1;

    f32x8 acc[2][NT];
    #pragma unroll
    for (int mt = 0; mt < 2; ++mt)
        #pragma unroll
        for (int nt = 0; nt < NT; ++nt)
            acc[mt][nt] = (f32x8)0.0f;

    #pragma unroll 1
    for (int pair = 0; pair < 2; ++pair) {
        const float*  __restrict__ A   = pair ? A1 : A0;
        const __bf16* __restrict__ Bhi = pair ? B1hi : B0hi;
        const __bf16* __restrict__ Blo = pair ? B1lo : B0lo;
        const int K = pair ? K1 : K0;
        if (K == 0) continue;

        const float*  aBase = A + (size_t)(blockM + aRow) * K + aCol;
        const __bf16* bBase = (bPl ? Blo : Bhi) + (size_t)(blockN + bRow) * K;

        auto stageB = [&](int kb, int buf) {
            if (!bActive) return;
            __bf16* dst = (bPl ? &sBlo[buf][bRow * LDS_STRIDE]
                               : &sBhi[buf][bRow * LDS_STRIDE]);
            async_copy_row64(bBase + kb, dst);
        };
        auto stageA = [&](int kb, int buf) {
            if (kb + BK < K)   // prefetch the tile after the one being staged
                __builtin_prefetch(aBase + kb + BK, 0, 0);
            #pragma unroll
            for (int j = 0; j < 8; ++j) {
                f32x2 xv = *(const f32x2*)(aBase + (size_t)16 * K * j + kb);
                __bf16 h0 = (__bf16)xv.x, h1 = (__bf16)xv.y;
                bf16x2 hp, lp;
                hp.x = h0;  hp.y = h1;
                lp.x = (__bf16)(xv.x - (float)h0);
                lp.y = (__bf16)(xv.y - (float)h1);
                int o = (aRow + 16 * j) * LDS_STRIDE + aCol;
                *(bf16x2*)&sAhi[buf][o] = hp;
                *(bf16x2*)&sAlo[buf][o] = lp;
            }
        };

        // fragment loader (ISA 16-bit 16x32 layout): lane l16 = row/col,
        // elems 0..7 = K[8*kh..], elems 8..15 = K[16+8*kh..]
        auto frag = [&](const __bf16* plane, int row) -> bf16x16 {
            const __bf16* p0 = plane + row * LDS_STRIDE + kh * 8;
            bf16x8 x0 = *(const bf16x8*)(p0);
            bf16x8 x1 = *(const bf16x8*)(p0 + 16);
            return __builtin_shufflevector(x0, x1,
                    0,1,2,3,4,5,6,7,8,9,10,11,12,13,14,15);
        };

        // one pipeline step: consume buf, optionally stage tile (s+1) -> buf^1
        auto step = [&](int s, int buf, bool doStage) {
            wait_async_lds();       // our async writes into buf
            __syncthreads();        // everyone's staging of buf done

            bf16x16 aHi[2], aLo[2], bHi[NT], bLo[NT];
            #pragma unroll
            for (int mt = 0; mt < 2; ++mt) {
                int m = waveM * 32 + mt * 16 + l16;
                aHi[mt] = frag(sAhi[buf], m);
                aLo[mt] = frag(sAlo[buf], m);
            }
            #pragma unroll
            for (int nt = 0; nt < NT; ++nt) {
                int n = waveN * (NT * 16) + nt * 16 + l16;
                bHi[nt] = frag(sBhi[buf], n);
                bLo[nt] = frag(sBlo[buf], n);
            }

            if (doStage) {          // overlaps with the WMMA burst below
                stageB((s + 1) * BK, buf ^ 1);
                stageA((s + 1) * BK, buf ^ 1);
            }

            #pragma unroll
            for (int mt = 0; mt < 2; ++mt) {
                #pragma unroll
                for (int nt = 0; nt < NT; ++nt) {
                    acc[mt][nt] = __builtin_amdgcn_wmma_f32_16x16x32_bf16(
                        false, aHi[mt], false, bHi[nt], (short)0, acc[mt][nt], false, false);
                    acc[mt][nt] = __builtin_amdgcn_wmma_f32_16x16x32_bf16(
                        false, aHi[mt], false, bLo[nt], (short)0, acc[mt][nt], false, false);
                    acc[mt][nt] = __builtin_amdgcn_wmma_f32_16x16x32_bf16(
                        false, aLo[mt], false, bHi[nt], (short)0, acc[mt][nt], false, false);
                }
            }
        };

        __syncthreads();            // protect buffer reuse across pairs
        stageB(0, 0);
        stageA(0, 0);

        // nsteps = K/32 is even for every layer (8/16/48/96)
        const int nsteps = K / BK;
        int s = 0;
        #pragma unroll 1
        for (; s + 2 < nsteps; s += 2) {
            step(s,     0, true);   // literal buffers: LDS offsets fold to imms
            step(s + 1, 1, true);
        }
        step(nsteps - 2, 0, true);  // peeled tail: stage last tile...
        step(nsteps - 1, 1, false); // ...then drain without staging
    }

    // ---- epilogue: C/D layout: lane l, vgpr v -> m = v + 8*(l>=16), n = l&15
    #pragma unroll
    for (int mt = 0; mt < 2; ++mt) {
        #pragma unroll
        for (int nt = 0; nt < NT; ++nt) {
            int n = blockN + waveN * (NT * 16) + nt * 16 + l16;
            float bv = bias ? bias[n] : 0.0f;
            #pragma unroll
            for (int v = 0; v < 8; ++v) {
                int m = blockM + waveM * 32 + mt * 16 + kh * 8 + v;
                float val = acc[mt][nt][v] + bv;
                if (do_relu) val = fmaxf(val, 0.0f);
                OUT[(size_t)m * ldo + n] = val;
            }
        }
    }
}

// ---------------------------------------------------------------------------
// Final layer: score[m] = h3[m, 0:64] . W4 + b4
// ---------------------------------------------------------------------------
__global__ __launch_bounds__(256)
void final_score(const float* __restrict__ h3, const float* __restrict__ W4,
                 const float* __restrict__ b4, float* __restrict__ out, int M)
{
    int r = blockIdx.x * 256 + threadIdx.x;
    if (r >= M) return;
    const float* row = h3 + (size_t)r * 64;
    float s = b4[0];
    #pragma unroll
    for (int i = 0; i < 64; ++i) s += row[i] * W4[i];
    out[r] = s;
}

// ---------------------------------------------------------------------------
// Launch
// ---------------------------------------------------------------------------
extern "C" void kernel_launch(void* const* d_in, const int* in_sizes, int n_in,
                              void* d_out, int out_size, void* d_ws, size_t ws_size,
                              hipStream_t stream)
{
    (void)in_sizes; (void)n_in; (void)out_size; (void)ws_size;

    const int B = 4096, E = 65536, D = 1536;

    const float* x_skill  = (const float*)d_in[0];
    const float* x_job    = (const float*)d_in[1];
    const float* x_resume = (const float*)d_in[2];
    const int*   src_sj   = (const int*)d_in[3];
    const int*   dst_sj   = (const int*)d_in[4];
    const int*   src_sr   = (const int*)d_in[5];
    const int*   dst_sr   = (const int*)d_in[6];
    const float* W_rel_j  = (const float*)d_in[7];
    const float* b_rel_j  = (const float*)d_in[8];
    const float* W_root_j = (const float*)d_in[9];
    const float* W_rel_r  = (const float*)d_in[10];
    const float* b_rel_r  = (const float*)d_in[11];
    const float* W_root_r = (const float*)d_in[12];
    const float* W1 = (const float*)d_in[13]; const float* b1 = (const float*)d_in[14];
    const float* W2 = (const float*)d_in[15]; const float* b2 = (const float*)d_in[16];
    const float* W3 = (const float*)d_in[17]; const float* b3 = (const float*)d_in[18];
    const float* W4 = (const float*)d_in[19]; const float* b4 = (const float*)d_in[20];

    // ---- workspace layout
    char* p = (char*)d_ws;
    auto alloc_f = [&](size_t n) { float*  r = (float*)p;  p += n * sizeof(float);  return r; };
    auto alloc_b = [&](size_t n) { __bf16* r = (__bf16*)p; p += n * sizeof(__bf16); return r; };

    float* agg_j = alloc_f((size_t)B * D);
    float* agg_r = alloc_f((size_t)B * D);
    float* h     = alloc_f((size_t)B * 2 * D);      // [h_job | h_res] concat in place
    float* m1    = alloc_f((size_t)B * 512);
    float* m2    = alloc_f((size_t)B * 256);
    float* m3    = alloc_f((size_t)B * 64);

    __bf16* wjrel_hi  = alloc_b((size_t)D * D); __bf16* wjrel_lo  = alloc_b((size_t)D * D);
    __bf16* wjroot_hi = alloc_b((size_t)D * D); __bf16* wjroot_lo = alloc_b((size_t)D * D);
    __bf16* wrrel_hi  = alloc_b((size_t)D * D); __bf16* wrrel_lo  = alloc_b((size_t)D * D);
    __bf16* wrroot_hi = alloc_b((size_t)D * D); __bf16* wrroot_lo = alloc_b((size_t)D * D);
    __bf16* w1_hi = alloc_b((size_t)2 * D * 512); __bf16* w1_lo = alloc_b((size_t)2 * D * 512);
    __bf16* w2_hi = alloc_b((size_t)512 * 256);   __bf16* w2_lo = alloc_b((size_t)512 * 256);
    __bf16* w3_hi = alloc_b((size_t)256 * 64);    __bf16* w3_lo = alloc_b((size_t)256 * 64);

    // ---- weight split/transpose (deterministic, every launch)
    auto prep = [&](const float* W, __bf16* hi, __bf16* lo, int K, int N) {
        int tot = K * N;
        prep_weight<<<(tot + 255) / 256, 256, 0, stream>>>(W, hi, lo, K, N);
    };
    prep(W_rel_j,  wjrel_hi,  wjrel_lo,  D, D);
    prep(W_root_j, wjroot_hi, wjroot_lo, D, D);
    prep(W_rel_r,  wrrel_hi,  wrrel_lo,  D, D);
    prep(W_root_r, wrroot_hi, wrroot_lo, D, D);
    prep(W1, w1_hi, w1_lo, 2 * D, 512);
    prep(W2, w2_hi, w2_lo, 512, 256);
    prep(W3, w3_hi, w3_lo, 256, 64);

    // ---- segment-sum via atomics
    hipMemsetAsync(agg_j, 0, (size_t)2 * B * D * sizeof(float), stream);
    gcn_scatter_add<<<E, 256, 0, stream>>>(x_skill, src_sj, dst_sj, agg_j, D);
    gcn_scatter_add<<<E, 256, 0, stream>>>(x_skill, src_sr, dst_sr, agg_r, D);

    // ---- GraphConv GEMMs (fused rel+root), outputs concat into h
    gemm_bf16split<4><<<dim3(B / BM, D / 128), 256, 0, stream>>>(
        agg_j, wjrel_hi, wjrel_lo, D, x_job, wjroot_hi, wjroot_lo, D,
        b_rel_j, h, 2 * D, 1);
    gemm_bf16split<4><<<dim3(B / BM, D / 128), 256, 0, stream>>>(
        agg_r, wrrel_hi, wrrel_lo, D, x_resume, wrroot_hi, wrroot_lo, D,
        b_rel_r, h + D, 2 * D, 1);

    // ---- MLP
    gemm_bf16split<4><<<dim3(B / BM, 512 / 128), 256, 0, stream>>>(
        h, w1_hi, w1_lo, 2 * D, nullptr, nullptr, nullptr, 0, b1, m1, 512, 1);
    gemm_bf16split<4><<<dim3(B / BM, 256 / 128), 256, 0, stream>>>(
        m1, w2_hi, w2_lo, 512, nullptr, nullptr, nullptr, 0, b2, m2, 256, 1);
    gemm_bf16split<2><<<dim3(B / BM, 64 / 64), 256, 0, stream>>>(
        m2, w3_hi, w3_lo, 256, nullptr, nullptr, nullptr, 0, b3, m3, 64, 1);

    final_score<<<(B + 255) / 256, 256, 0, stream>>>(m3, W4, b4, (float*)d_out, B);
}
